// BrainGNNBlock_81784767250574
// MI455X (gfx1250) — compile-verified, matrix operand order checked by reading, not run on hardware
//
#include <hip/hip_runtime.h>
#include <hip/hip_bf16.h>
#include <math.h>

// ---------------------------------------------------------------------------
// BrainGNN block for MI455X (gfx1250, wave32, WMMA).
//   B=256 graphs, NR=268 ROIs, N=68608 nodes, IC=256, OC=64, NC=7,
//   DEG=32, E=N*32, K=214 kept nodes per graph.
// ---------------------------------------------------------------------------

#define Bc   256
#define NRc  268
#define Nc   (Bc * NRc)       // 68608
#define ICc  256
#define OCc  64
#define NCc  7
#define DEGc 32
#define Ec   (Nc * DEGc)      // 2195456
#define Kc   214
#define LN_EPS 1e-5f

typedef __attribute__((ext_vector_type(16))) _Float16 v16h;
typedef __attribute__((ext_vector_type(8)))  _Float16 v8h;
typedef __attribute__((ext_vector_type(8)))  float    v8f;

static __device__ __forceinline__ v16h cat16(v8h lo, v8h hi) {
  v16h r;
#pragma unroll
  for (int e = 0; e < 8; ++e) { r[e] = lo[e]; r[e + 8] = hi[e]; }
  return r;
}

static __device__ __forceinline__ float sigmoidf(float z) {
  return 1.0f / (1.0f + __expf(-z));
}

// ---------------------------------------------------------------------------
// K0: zero a float buffer (graph-capture-safe replacement for hipMemsetAsync)
// ---------------------------------------------------------------------------
__global__ __launch_bounds__(256) void k_zero(float4* __restrict__ p, int n4) {
  int i = blockIdx.x * 256 + threadIdx.x;
  if (i < n4) p[i] = make_float4(0.f, 0.f, 0.f, 0.f);
}

// ---------------------------------------------------------------------------
// K1: cw = softmax(roi_community, axis=-1)   (268 x 7)
// ---------------------------------------------------------------------------
__global__ __launch_bounds__(256) void k_softmax(const float* __restrict__ rc,
                                                 float* __restrict__ cw) {
  int r = blockIdx.x * 256 + threadIdx.x;
  if (r >= NRc) return;
  float v[NCc], m = -1e30f;
#pragma unroll
  for (int c = 0; c < NCc; ++c) { v[c] = rc[r * NCc + c]; m = fmaxf(m, v[c]); }
  float s = 0.f;
#pragma unroll
  for (int c = 0; c < NCc; ++c) { v[c] = __expf(v[c] - m); s += v[c]; }
  float inv = 1.0f / s;
#pragma unroll
  for (int c = 0; c < NCc; ++c) cw[r * NCc + c] = v[c] * inv;
}

// ---------------------------------------------------------------------------
// K2: roi_k[r,i,o] = sum_c cw[r,c] * bk[c,i,o], stored as f16 for WMMA B-operand
// grid = 268 blocks x 256 threads (thread = input channel i)
// ---------------------------------------------------------------------------
__global__ __launch_bounds__(256) void k_roik(const float* __restrict__ cw,
                                              const float* __restrict__ bk,
                                              _Float16* __restrict__ rkh) {
  const int r = blockIdx.x;
  const int i = threadIdx.x;
  float c[NCc];
#pragma unroll
  for (int q = 0; q < NCc; ++q) c[q] = cw[r * NCc + q];
  float acc[OCc];
#pragma unroll
  for (int o = 0; o < OCc; ++o) acc[o] = 0.f;
#pragma unroll
  for (int q = 0; q < NCc; ++q) {
    const float* src = &bk[((size_t)q * ICc + i) * OCc];
#pragma unroll
    for (int o = 0; o < OCc; o += 4) {
      float4 f = *(const float4*)&src[o];
      acc[o + 0] += c[q] * f.x;
      acc[o + 1] += c[q] * f.y;
      acc[o + 2] += c[q] * f.z;
      acc[o + 3] += c[q] * f.w;
    }
  }
  _Float16* dst = &rkh[((size_t)r * ICc + i) * OCc];
#pragma unroll
  for (int o = 0; o < OCc; ++o) dst[o] = (_Float16)acc[o];
}

// ---------------------------------------------------------------------------
// K3: per-ROI GEMM via WMMA.  For each r: C[256x64] = A[256x256] x B[256x64]
//   A row b = x[b*NR + r, :] (f32 -> f16 in LDS), B = roi_k[r] (f16, transposed
//   to [n][k] in LDS so B fragments are contiguous-K 16B ds loads).
// One workgroup (8 waves) per r; wave w owns rows [32w,32w+32), all 64 cols.
// ---------------------------------------------------------------------------
__global__ __launch_bounds__(256) void k_gemm(const float* __restrict__ x,
                                              const _Float16* __restrict__ rkh,
                                              float* __restrict__ xt) {
  const int r    = blockIdx.x;
  const int tid  = threadIdx.x;
  const int lane = tid & 31;
  const int wv   = tid >> 5;

  __shared__ __align__(16) _Float16 lA[256 * 40];  // 256 rows x 32K, stride 40
  __shared__ __align__(16) _Float16 lB[64 * 40];   // Bt: 64 cols x 32K, stride 40

  v8f acc[2][4];
#pragma unroll
  for (int mt = 0; mt < 2; ++mt)
#pragma unroll
    for (int nt = 0; nt < 4; ++nt)
#pragma unroll
      for (int e = 0; e < 8; ++e) acc[mt][nt][e] = 0.f;

  const size_t xrow = ((size_t)tid * NRc + r) * ICc;  // thread tid stages row b=tid
  const int kB = tid & 31, n8 = tid >> 5;             // B-staging role

  const int half = lane >> 4, lm = lane & 15, base = half * 8;

  for (int kc = 0; kc < 8; ++kc) {
    const int k0 = kc * 32;
    if (kc < 7) __builtin_prefetch(&x[xrow + k0 + 32], 0, 3);  // global_prefetch_b8
    // ---- stage A chunk: 256 rows x 32 K, f32 -> f16
#pragma unroll
    for (int j = 0; j < 8; ++j) {
      float4 f = *(const float4*)&x[xrow + k0 + j * 4];
      lA[tid * 40 + j * 4 + 0] = (_Float16)f.x;
      lA[tid * 40 + j * 4 + 1] = (_Float16)f.y;
      lA[tid * 40 + j * 4 + 2] = (_Float16)f.z;
      lA[tid * 40 + j * 4 + 3] = (_Float16)f.w;
    }
    // ---- stage B chunk transposed: lB[n][k] = rkh[r][k0+k][n]
    {
      v8h bv = *(const v8h*)&rkh[((size_t)r * ICc + (k0 + kB)) * OCc + n8 * 8];
#pragma unroll
      for (int j = 0; j < 8; ++j) lB[(n8 * 8 + j) * 40 + kB] = bv[j];
    }
    __syncthreads();
    // ---- compute: 2 M-tiles x 4 N-tiles, one K=32 WMMA step each
#pragma unroll
    for (int mt = 0; mt < 2; ++mt) {
      const int row = wv * 32 + mt * 16 + lm;
      v8h alo = *(const v8h*)&lA[row * 40 + base];
      v8h ahi = *(const v8h*)&lA[row * 40 + base + 16];
      v16h a = cat16(alo, ahi);
#pragma unroll
      for (int nt = 0; nt < 4; ++nt) {
        const int col = nt * 16 + lm;
        v8h blo = *(const v8h*)&lB[col * 40 + base];
        v8h bhi = *(const v8h*)&lB[col * 40 + base + 16];
        v16h b = cat16(blo, bhi);
        acc[mt][nt] = __builtin_amdgcn_wmma_f32_16x16x32_f16(
            false, a, false, b, (short)0, acc[mt][nt], false, false);
      }
    }
    __syncthreads();
  }

  // ---- epilogue: C layout -> xt[(b*NR+r)*64 + n]
#pragma unroll
  for (int mt = 0; mt < 2; ++mt)
#pragma unroll
    for (int nt = 0; nt < 4; ++nt)
#pragma unroll
      for (int v = 0; v < 8; ++v) {
        int b   = wv * 32 + mt * 16 + v + half * 8;
        int col = nt * 16 + lm;
        xt[((size_t)b * NRc + r) * OCc + col] = acc[mt][nt][v];
      }
}

// ---------------------------------------------------------------------------
// K4: edge-gated message scatter: agg[dst] += xt[src] * sigmoid(ea*w + b)
// 8 threads per edge, 8 channels each; f32 atomics stay resident in 192MB L2.
// ---------------------------------------------------------------------------
__global__ __launch_bounds__(256) void k_edge(const int* __restrict__ ei,
                                              const float* __restrict__ ea,
                                              const float* __restrict__ xt,
                                              const float* __restrict__ eww,
                                              const float* __restrict__ ewb,
                                              float* __restrict__ agg, int E) {
  int t  = blockIdx.x * 256 + threadIdx.x;
  int e  = t >> 3;
  int cg = t & 7;
  if (e >= E) return;
  int   src = ei[e];
  int   dst = ei[E + e];
  float a   = ea[e];
  int   o0  = cg * 8;
  const float* xs = &xt[(size_t)src * OCc + o0];
  float*       dp = &agg[(size_t)dst * OCc + o0];
#pragma unroll
  for (int j = 0; j < 8; ++j) {
    float g = sigmoidf(a * eww[o0 + j] + ewb[o0 + j]);
    atomicAdd(&dp[j], xs[j] * g);  // global_atomic_add_f32
  }
}

// ---------------------------------------------------------------------------
// K5: fused self-loop (gate const per channel) + conv bias + ELU + LayerNorm.
// One thread per node, in-place: agg -> h.
// ---------------------------------------------------------------------------
__global__ __launch_bounds__(256) void k_node(float* __restrict__ agg,
                                              const float* __restrict__ xt,
                                              const float* __restrict__ eww,
                                              const float* __restrict__ ewb,
                                              const float* __restrict__ cb,
                                              const float* __restrict__ gamma,
                                              const float* __restrict__ beta) {
  int n = blockIdx.x * 256 + threadIdx.x;  // N = 268*256 exactly
  float*       ap = &agg[(size_t)n * OCc];
  const float* xp = &xt[(size_t)n * OCc];
  float v[OCc];
  float s = 0.f;
#pragma unroll
  for (int o = 0; o < OCc; ++o) {
    float g = sigmoidf(eww[o] + ewb[o]);      // self-loop edge_attr == 1.0
    float t = ap[o] + xp[o] * g + cb[o];
    t = (t > 0.f) ? t : (__expf(t) - 1.f);    // ELU
    v[o] = t;
    s += t;
  }
  float mu  = s * (1.0f / OCc);
  float var = 0.f;
#pragma unroll
  for (int o = 0; o < OCc; ++o) { float d = v[o] - mu; var += d * d; }
  var *= (1.0f / OCc);
  float inv = __frsqrt_rn(var + LN_EPS);
#pragma unroll
  for (int o = 0; o < OCc; ++o) ap[o] = (v[o] - mu) * inv * gamma[o] + beta[o];
}

// ---------------------------------------------------------------------------
// K6: attention scores via WMMA: s = tanh(h @ W1 + b1) @ W2 + b2.
// 8 waves/block, 16 nodes per wave (128 nodes/block, N = 536*128 exactly).
// W1 staged transposed in LDS once; K=64 -> two WMMA k-steps.
// ---------------------------------------------------------------------------
__global__ __launch_bounds__(256) void k_att(const float* __restrict__ h,
                                             const float* __restrict__ w1,
                                             const float* __restrict__ b1,
                                             const float* __restrict__ w2,
                                             const float* __restrict__ b2,
                                             float* __restrict__ scores) {
  const int tid  = threadIdx.x;
  const int lane = tid & 31;
  const int wv   = tid >> 5;
  const int nodebase = blockIdx.x * 128;

  __shared__ __align__(16) _Float16 lW[64 * 72];   // Wt[o][i]
  __shared__ __align__(16) _Float16 lH[128 * 72];  // h tile, f16

  for (int idx = tid; idx < 64 * 64; idx += 256) {
    int i = idx >> 6, o = idx & 63;
    lW[o * 72 + i] = (_Float16)w1[idx];
  }
  {
    int nl = tid >> 1, hs = tid & 1;
    const float* src = &h[((size_t)(nodebase + nl)) * OCc + hs * 32];
#pragma unroll
    for (int j = 0; j < 32; ++j) lH[nl * 72 + hs * 32 + j] = (_Float16)src[j];
  }
  __syncthreads();

  const int half = lane >> 4, lm = lane & 15, base = half * 8;
  v8f acc[4];
#pragma unroll
  for (int nt = 0; nt < 4; ++nt) {
    float bv = b1[nt * 16 + lm];
#pragma unroll
    for (int e = 0; e < 8; ++e) acc[nt][e] = bv;
  }
  const int row = wv * 16 + lm;
#pragma unroll
  for (int kc = 0; kc < 2; ++kc) {
    v8h alo = *(const v8h*)&lH[row * 72 + kc * 32 + base];
    v8h ahi = *(const v8h*)&lH[row * 72 + kc * 32 + base + 16];
    v16h a = cat16(alo, ahi);
#pragma unroll
    for (int nt = 0; nt < 4; ++nt) {
      int col = nt * 16 + lm;
      v8h blo = *(const v8h*)&lW[col * 72 + kc * 32 + base];
      v8h bhi = *(const v8h*)&lW[col * 72 + kc * 32 + base + 16];
      v16h b = cat16(blo, bhi);
      acc[nt] = __builtin_amdgcn_wmma_f32_16x16x32_f16(
          false, a, false, b, (short)0, acc[nt], false, false);
    }
  }
  // p[v] = sum_n tanh(t[m][n]) * w2[n]; n = nt*16 + lm spans lanes of a half
  float p[8];
#pragma unroll
  for (int v = 0; v < 8; ++v) p[v] = 0.f;
#pragma unroll
  for (int nt = 0; nt < 4; ++nt) {
    float wvv = w2[nt * 16 + lm];
#pragma unroll
    for (int v = 0; v < 8; ++v) p[v] += tanhf(acc[nt][v]) * wvv;
  }
#pragma unroll
  for (int off = 1; off < 16; off <<= 1)
#pragma unroll
    for (int v = 0; v < 8; ++v) p[v] += __shfl_xor(p[v], off, 32);
  if (lm == 0) {
    float bb = b2[0];
#pragma unroll
    for (int v = 0; v < 8; ++v)
      scores[nodebase + wv * 16 + half * 8 + v] = p[v] + bb;
  }
}

// ---------------------------------------------------------------------------
// K7: per-graph top-K (rank-based, matches lax.top_k descending + stable ties)
// + gather pooled output. One block per graph.
// ---------------------------------------------------------------------------
__global__ __launch_bounds__(256) void k_topk(const float* __restrict__ h,
                                              const float* __restrict__ scores,
                                              float* __restrict__ xp,
                                              float* __restrict__ bp,
                                              float* __restrict__ pm) {
  const int b = blockIdx.x, tid = threadIdx.x;
  __shared__ float s[NRc];
  for (int i = tid; i < NRc; i += 256) s[i] = scores[b * NRc + i];
  __syncthreads();
  for (int i = tid; i < NRc; i += 256) {
    float si = s[i];
    int rank = 0;
    for (int j = 0; j < NRc; ++j) {
      float sj = s[j];
      rank += (sj > si) || (sj == si && j < i);
    }
    if (rank < Kc) {
      int outrow = b * Kc + rank;
      pm[outrow] = (float)(b * NRc + i);
      bp[outrow] = (float)b;
      float g = sigmoidf(si);
      const float* hv = &h[((size_t)(b * NRc + i)) * OCc];
      float*       ov = &xp[(size_t)outrow * OCc];
#pragma unroll
      for (int o = 0; o < OCc; o += 4) {
        float4 f = *(const float4*)&hv[o];
        f.x *= g; f.y *= g; f.z *= g; f.w *= g;
        *(float4*)&ov[o] = f;
      }
    }
  }
}

// ---------------------------------------------------------------------------
extern "C" void kernel_launch(void* const* d_in, const int* in_sizes, int n_in,
                              void* d_out, int out_size, void* d_ws, size_t ws_size,
                              hipStream_t stream) {
  const float* x   = (const float*)d_in[0];
  const int*   ei  = (const int*)  d_in[1];
  const float* ea  = (const float*)d_in[2];
  // d_in[3] = batch (implicit from layout, unused)
  const float* bk  = (const float*)d_in[4];
  const float* rc  = (const float*)d_in[5];
  const float* eww = (const float*)d_in[6];
  const float* ewb = (const float*)d_in[7];
  const float* cb  = (const float*)d_in[8];
  const float* lng = (const float*)d_in[9];
  const float* lnb = (const float*)d_in[10];
  const float* w1  = (const float*)d_in[11];
  const float* b1  = (const float*)d_in[12];
  const float* w2  = (const float*)d_in[13];
  const float* b2  = (const float*)d_in[14];
  const int E = in_sizes[1] / 2;

  // workspace layout (256B-aligned blocks)
  char* w = (char*)d_ws;
  float*     cw  = (float*)w;                              // 268*7 f32
  _Float16*  rkh = (_Float16*)(w + 7680);                  // 268*256*64 f16
  float*     xt  = (float*)(w + 7680 + 8781824);           // N*64 f32
  float*     agg = (float*)(w + 7680 + 8781824 + 17563648);// N*64 f32 (-> h in-place)

  // d_out layout: [x_pooled (B*K*64)] [batch_pooled (B*K)] [scores (N)] [perm (B*K)]
  float* xp = (float*)d_out;
  float* bp = xp + (size_t)Bc * Kc * OCc;
  float* sc = bp + (size_t)Bc * Kc;
  float* pm = sc + (size_t)Nc;

  k_softmax<<<2, 256, 0, stream>>>(rc, cw);
  k_roik  <<<NRc, 256, 0, stream>>>(cw, bk, rkh);
  k_zero  <<<(Nc * OCc / 4 + 255) / 256, 256, 0, stream>>>((float4*)agg, Nc * OCc / 4);
  k_gemm  <<<NRc, 256, 0, stream>>>(x, rkh, xt);
  k_edge  <<<((size_t)E * 8 + 255) / 256, 256, 0, stream>>>(ei, ea, xt, eww, ewb, agg, E);
  k_node  <<<Nc / 256, 256, 0, stream>>>(agg, xt, eww, ewb, cb, lng, lnb);
  k_att   <<<Nc / 128, 256, 0, stream>>>(agg, w1, b1, w2, b2, sc);
  k_topk  <<<Bc, 256, 0, stream>>>(agg, sc, xp, bp, pm);
}
